// AttentionFusionLayer_20899310862505
// MI455X (gfx1250) — compile-verified, matrix operand order checked by reading, not run on hardware
//
#include <hip/hip_runtime.h>

// ---------------------------------------------------------------------------
// AttentionFusionLayer for MI455X (gfx1250, wave32, WMMA + async-to-LDS)
// Pipeline: f32->bf16 converts, 6 WMMA GEMMs + scores GEMM + softmax,
// residual fused into attn GEMM epilogue, final o-proj writes f32 output.
// ---------------------------------------------------------------------------

typedef __attribute__((ext_vector_type(16))) __bf16 v16bf;
typedef __attribute__((ext_vector_type(8)))  __bf16 v8bf;
typedef __attribute__((ext_vector_type(8)))  float  v8f;
typedef int vsi4 __attribute__((__vector_size__(4 * sizeof(int))));

union FragBF { v16bf v; v8bf h[2]; };

#if __has_builtin(__builtin_amdgcn_global_load_async_to_lds_b128)
#define USE_ASYNC_LDS 1
#else
#define USE_ASYNC_LDS 0
#endif

__device__ __forceinline__ __bf16 f2bf(float f) {
  unsigned u = __float_as_uint(f);
  u += 0x7FFFu + ((u >> 16) & 1u);           // round-to-nearest-even
  unsigned short s = (unsigned short)(u >> 16);
  return __builtin_bit_cast(__bf16, s);
}
__device__ __forceinline__ float bf2f(__bf16 b) {
  unsigned short s = __builtin_bit_cast(unsigned short, b);
  return __uint_as_float(((unsigned)s) << 16);
}

#if USE_ASYNC_LDS
__device__ __forceinline__ void async_b128(const __bf16* g, __bf16* l) {
  __builtin_amdgcn_global_load_async_to_lds_b128((vsi4*)g, (vsi4*)l, 0, 0);
}
__device__ __forceinline__ void wait_async0() {
#if __has_builtin(__builtin_amdgcn_s_wait_asynccnt)
  __builtin_amdgcn_s_wait_asynccnt(0);
#else
  asm volatile("s_wait_asynccnt 0" ::: "memory");
#endif
}
#endif

// ---------------------------------------------------------------------------
// f32 -> bf16 conversion (vectorized)
// ---------------------------------------------------------------------------
__global__ __launch_bounds__(256)
void cvt_f32_bf16(const float* __restrict__ x, __bf16* __restrict__ y, long n) {
  long i = ((long)blockIdx.x * 256 + threadIdx.x) * 4;
  if (i + 3 < n) {
    float4 f = *(const float4*)(x + i);
    y[i + 0] = f2bf(f.x);
    y[i + 1] = f2bf(f.y);
    y[i + 2] = f2bf(f.z);
    y[i + 3] = f2bf(f.w);
  } else {
    for (; i < n; ++i) y[i] = f2bf(x[i]);
  }
}

// ---------------------------------------------------------------------------
// Generic WMMA GEMM:  C[m,n] = scale * sum_k A[m,k] * W[n,k]  (+bias[n]) (+resid[m,n])
// A: bf16 [M x K] rows lda, batch stride sAb
// W: bf16 [N x K] rows ldw, batch stride sWb   (nn.Linear weight layout)
// Outputs (any may be null): outF (f32), outB (bf16), outBT (bf16 transposed:
//   outBT[n*ldoT + m]).
// Block tile 128x128x32; 256 threads = 8 waves in 2(M) x 4(N); each wave owns a
// 64x32 sub-tile = 4x2 C fragments -> 8x v_wmma_f32_16x16x32_bf16 per K-step.
// Tile feed: GLOBAL_LOAD_ASYNC_TO_LDS_B128 (ASYNCcnt) when available, double
// buffered with one barrier per K-step; LDS rows padded to 80B (conflict-free).
// ---------------------------------------------------------------------------
__global__ __launch_bounds__(256)
void wmma_gemm_kernel(const __bf16* __restrict__ A, long sAb, int lda,
                      const __bf16* __restrict__ W, long sWb, int ldw,
                      const float* __restrict__ bias,
                      const __bf16* __restrict__ resid, long sRb,
                      float* __restrict__ outF,
                      __bf16* __restrict__ outB, long sOb, int ldo,
                      __bf16* __restrict__ outBT, long sOTb, int ldoT,
                      int K, float scale) {
  __shared__ __bf16 sA[2][128][40];
  __shared__ __bf16 sB[2][128][40];

  const int bz   = blockIdx.z;
  const int m0   = blockIdx.y * 128;
  const int n0   = blockIdx.x * 128;
  const int tid  = threadIdx.x;
  const int wave = tid >> 5;
  const int lane = tid & 31;
  const int wm   = (wave >> 2) * 64;   // wave's M offset within block tile
  const int wn   = (wave & 3) * 32;    // wave's N offset within block tile
  const int fr   = lane & 15;          // fragment row/col within 16
  const int fh   = lane >> 4;          // half select

  const int lr = tid >> 2;
  const int lc = (tid & 3) * 8;
  const __bf16* Ab = A + bz * sAb + (long)(m0 + lr) * lda + lc;
  const __bf16* Wb = W + bz * sWb + (long)(n0 + lr) * ldw + lc;

  auto issueTile = [&](int kt, int buf) {
    long off = (long)kt * 32;
#if USE_ASYNC_LDS
    async_b128(Ab + off,             &sA[buf][lr][lc]);
    async_b128(Ab + off + 64L * lda, &sA[buf][lr + 64][lc]);
    async_b128(Wb + off,             &sB[buf][lr][lc]);
    async_b128(Wb + off + 64L * ldw, &sB[buf][lr + 64][lc]);
#else
    v8bf ra0 = *(const v8bf*)(Ab + off);
    v8bf ra1 = *(const v8bf*)(Ab + off + 64L * lda);
    v8bf rb0 = *(const v8bf*)(Wb + off);
    v8bf rb1 = *(const v8bf*)(Wb + off + 64L * ldw);
    *(v8bf*)&sA[buf][lr][lc]      = ra0;
    *(v8bf*)&sA[buf][lr + 64][lc] = ra1;
    *(v8bf*)&sB[buf][lr][lc]      = rb0;
    *(v8bf*)&sB[buf][lr + 64][lc] = rb1;
#endif
  };

  v8f acc[4][2] = {};

  const int NT = K >> 5;   // K / 32 steps
  issueTile(0, 0);

  for (int kt = 0; kt < NT; ++kt) {
#if USE_ASYNC_LDS
    wait_async0();               // my async copies into buf cur have landed
#endif
    __syncthreads();             // all waves' copies landed; buf nxt free of readers
    if (kt + 1 < NT) issueTile(kt + 1, (kt + 1) & 1);  // overlaps compute below
    const int cur = kt & 1;

    FragBF af[4], bfrag[2];
#pragma unroll
    for (int mt = 0; mt < 4; ++mt) {
      const __bf16* p = &sA[cur][wm + mt * 16 + fr][fh * 8];
      af[mt].h[0] = *(const v8bf*)p;
      af[mt].h[1] = *(const v8bf*)(p + 16);
    }
#pragma unroll
    for (int nt = 0; nt < 2; ++nt) {
      const __bf16* p = &sB[cur][wn + nt * 16 + fr][fh * 8];
      bfrag[nt].h[0] = *(const v8bf*)p;
      bfrag[nt].h[1] = *(const v8bf*)(p + 16);
    }
#pragma unroll
    for (int mt = 0; mt < 4; ++mt)
#pragma unroll
      for (int nt = 0; nt < 2; ++nt)
        acc[mt][nt] = __builtin_amdgcn_wmma_f32_16x16x32_bf16(
            false, af[mt].v, false, bfrag[nt].v,
            (short)0, acc[mt][nt], false, false);
  }

  // Epilogue: scale, bias, residual, stores
#pragma unroll
  for (int mt = 0; mt < 4; ++mt) {
#pragma unroll
    for (int nt = 0; nt < 2; ++nt) {
#pragma unroll
      for (int r = 0; r < 8; ++r) {
        long row = m0 + wm + mt * 16 + fh * 8 + r;
        long col = n0 + wn + nt * 16 + fr;
        float val = acc[mt][nt][r] * scale;
        if (bias)  val += bias[col];
        if (resid) val += bf2f(resid[bz * sRb + row * ldo + col]);
        if (outF)  outF[bz * sOb + row * ldo + col] = val;
        if (outB)  outB[bz * sOb + row * ldo + col] = f2bf(val);
        if (outBT) outBT[bz * sOTb + col * (long)ldoT + row] = f2bf(val);
      }
    }
  }
}

// ---------------------------------------------------------------------------
// Row softmax: f32 scores [rows x n] -> bf16 probabilities. One WG per row.
// ---------------------------------------------------------------------------
__global__ __launch_bounds__(256)
void softmax_kernel(const float* __restrict__ S, __bf16* __restrict__ P, int n) {
  const long row = blockIdx.x;
  const float* s = S + row * (long)n;
  __bf16* p = P + row * (long)n;
  const int tid = threadIdx.x;
  __shared__ float red[8];

  float lmax = -3.4e38f;
  for (int i = tid; i < n; i += 256) lmax = fmaxf(lmax, s[i]);
#pragma unroll
  for (int o = 16; o > 0; o >>= 1) lmax = fmaxf(lmax, __shfl_xor(lmax, o, 32));
  if ((tid & 31) == 0) red[tid >> 5] = lmax;
  __syncthreads();
  float m = red[0];
#pragma unroll
  for (int i = 1; i < 8; ++i) m = fmaxf(m, red[i]);
  __syncthreads();

  float lsum = 0.f;
  for (int i = tid; i < n; i += 256) lsum += __expf(s[i] - m);
#pragma unroll
  for (int o = 16; o > 0; o >>= 1) lsum += __shfl_xor(lsum, o, 32);
  if ((tid & 31) == 0) red[tid >> 5] = lsum;
  __syncthreads();
  float sum = 0.f;
#pragma unroll
  for (int i = 0; i < 8; ++i) sum += red[i];
  const float inv = 1.0f / sum;

  for (int i = tid; i < n; i += 256) p[i] = f2bf(__expf(s[i] - m) * inv);
}

// ---------------------------------------------------------------------------
// Host launch
// ---------------------------------------------------------------------------
extern "C" void kernel_launch(void* const* d_in, const int* in_sizes, int n_in,
                              void* d_out, int out_size, void* d_ws, size_t ws_size,
                              hipStream_t stream) {
  (void)in_sizes; (void)n_in; (void)out_size; (void)ws_size;

  const long Bn = 8, L = 2048, F = 1024, Mall = Bn * L;  // 16384

  const float* hist = (const float*)d_in[0];
  const float* comb = (const float*)d_in[1];
  const float* hW = (const float*)d_in[2];  const float* hb = (const float*)d_in[3];
  const float* cW = (const float*)d_in[4];  const float* cb = (const float*)d_in[5];
  const float* qW = (const float*)d_in[6];  const float* qb = (const float*)d_in[7];
  const float* kW = (const float*)d_in[8];  const float* kb = (const float*)d_in[9];
  const float* vW = (const float*)d_in[10]; const float* vb = (const float*)d_in[11];
  const float* oW = (const float*)d_in[12]; const float* ob = (const float*)d_in[13];
  float* out = (float*)d_out;

  char* wsp = (char*)d_ws;
  auto take = [&](size_t bytes) -> char* {
    char* p = wsp;
    wsp += (bytes + 255) & ~(size_t)255;
    return p;
  };
  const size_t szActB = (size_t)Mall * F * 2;   // 33.5 MB bf16 activation
  const size_t szWB   = (size_t)F * F * 2;      // 2 MB bf16 weight

  __bf16* histB  = (__bf16*)take(szActB);
  __bf16* combB  = (__bf16*)take(szActB);
  __bf16* hWb    = (__bf16*)take(szWB);
  __bf16* cWb    = (__bf16*)take(szWB);
  __bf16* qWb    = (__bf16*)take(szWB);
  __bf16* kWb    = (__bf16*)take(szWB);
  __bf16* vWb    = (__bf16*)take(szWB);
  __bf16* oWb    = (__bf16*)take(szWB);
  __bf16* hB     = (__bf16*)take(szActB);
  __bf16* cB     = (__bf16*)take(szActB);
  __bf16* qB     = (__bf16*)take(szActB);
  __bf16* kB     = (__bf16*)take(szActB);
  __bf16* vT     = (__bf16*)take(szActB);       // [F x Mall] transposed v
  __bf16* attres = (__bf16*)take(szActB);       // attn_out + q, bf16
  float*  Sc     = (float*) take((size_t)Bn * L * L * 4);  // f32 scores
  __bf16* Pp     = (__bf16*)take((size_t)Bn * L * L * 2);  // bf16 probs

  auto cvt = [&](const float* x, __bf16* y, long n) {
    unsigned blocks = (unsigned)((n + 1023) / 1024);
    cvt_f32_bf16<<<dim3(blocks), dim3(256), 0, stream>>>(x, y, n);
  };
  cvt(hist, histB, Mall * F);
  cvt(comb, combB, Mall * F);
  cvt(hW, hWb, F * F);
  cvt(cW, cWb, F * F);
  cvt(qW, qWb, F * F);
  cvt(kW, kWb, F * F);
  cvt(vW, vWb, F * F);
  cvt(oW, oWb, F * F);

  auto gemm = [&](const __bf16* A, long sAb, int lda,
                  const __bf16* W, long sWb, int ldw,
                  const float* bias,
                  const __bf16* resid, long sRb,
                  float* oF, __bf16* oB, long sOb, int ldo,
                  __bf16* oBT, long sOTb, int ldoT,
                  long M, long N, int K, float scale, int batches) {
    dim3 g((unsigned)(N / 128), (unsigned)(M / 128), (unsigned)batches);
    wmma_gemm_kernel<<<g, dim3(256), 0, stream>>>(A, sAb, lda, W, sWb, ldw, bias,
                                                  resid, sRb, oF, oB, sOb, ldo,
                                                  oBT, sOTb, ldoT, K, scale);
  };

  const float iscale = 0.03125f;  // 1/sqrt(1024)

  // h = hist @ hW^T + hb            (bf16 out)
  gemm(histB, 0, F, hWb, 0, F, hb, nullptr, 0,
       nullptr, hB, 0, F, nullptr, 0, 0, Mall, F, F, 1.0f, 1);
  // c = comb @ cW^T + cb
  gemm(combB, 0, F, cWb, 0, F, cb, nullptr, 0,
       nullptr, cB, 0, F, nullptr, 0, 0, Mall, F, F, 1.0f, 1);
  // q = h @ qW^T + qb
  gemm(hB, 0, F, qWb, 0, F, qb, nullptr, 0,
       nullptr, qB, 0, F, nullptr, 0, 0, Mall, F, F, 1.0f, 1);
  // k = c @ kW^T + kb
  gemm(cB, 0, F, kWb, 0, F, kb, nullptr, 0,
       nullptr, kB, 0, F, nullptr, 0, 0, Mall, F, F, 1.0f, 1);
  // v = c @ vW^T + vb  -> stored transposed as vT[f, m]
  gemm(cB, 0, F, vWb, 0, F, vb, nullptr, 0,
       nullptr, nullptr, 0, F, vT, 0, (int)Mall, Mall, F, F, 1.0f, 1);
  // scores = (q @ k^T) * 1/sqrt(F)   per batch, f32 out
  gemm(qB, L * F, F, kB, L * F, F, nullptr, nullptr, 0,
       Sc, nullptr, L * L, (int)L, nullptr, 0, 0, L, L, (int)F, iscale, (int)Bn);
  // softmax rows -> bf16 probabilities
  softmax_kernel<<<dim3((unsigned)Mall), dim3(256), 0, stream>>>(Sc, Pp, (int)L);
  // attn_out = P @ v  (+ residual q), bf16 out; v consumed via vT as [N x K]
  gemm(Pp, L * L, (int)L, vT, L, (int)Mall, nullptr, qB, L * F,
       nullptr, attres, L * F, (int)F, nullptr, 0, 0, L, F, (int)L, 1.0f, (int)Bn);
  // out = (attn_out + q) @ oW^T + ob   (f32 -> d_out)
  gemm(attres, 0, F, oWb, 0, F, ob, nullptr, 0,
       out, nullptr, 0, F, nullptr, 0, 0, Mall, F, F, 1.0f, 1);
}
